// LSTM_RNN_Network_63668595195986
// MI455X (gfx1250) — matrix operand order repaired and checked by reading, not compiled, not run on hardware
//
#include <hip/hip_runtime.h>
#include <stdint.h>

// ---------------------------------------------------------------------------
// 2-layer LSTM for MI455X (gfx1250, wave32, WMMA bf16).
// Strategy:
//   * precompute x@W+b as big time-parallel WMMA GEMMs
//   * one persistent kernel per layer for the recurrence: 16 WGs loop over all
//     512 timesteps with an atomic grid barrier per step (launch overhead gone),
//     cell state c lives in registers for the whole sequence, U is L2-resident.
// ---------------------------------------------------------------------------

static constexpr int BATCH = 64;
static constexpr int TSEQ  = 512;
static constexpr int EMBD  = 512;
static constexpr int HID   = 1024;
static constexpr int G4    = 4096;            // 4*HID
static constexpr int MROWS = BATCH * TSEQ;    // 32768
static constexpr int NWG   = HID / 64;        // 16 persistent workgroups

typedef __attribute__((ext_vector_type(16))) __bf16 bf16x16;
typedef __attribute__((ext_vector_type(8)))  float  f32x8;

#define WMMA_BF16(A_, B_, C_) \
  __builtin_amdgcn_wmma_f32_16x16x32_bf16(false, (A_), false, (B_), (short)0, (C_), false, false)

// ---- scalar helpers --------------------------------------------------------

__device__ __forceinline__ __bf16 f2bf(float f) {
  union { float f; unsigned u; } x; x.f = f;
  unsigned r = (x.u + 0x7FFFu + ((x.u >> 16) & 1u)) >> 16;
  union { unsigned short s; __bf16 b; } y; y.s = (unsigned short)r;
  return y.b;
}
__device__ __forceinline__ float bf2f(__bf16 b) {
  union { unsigned short s; __bf16 b; } y; y.b = b;
  union { unsigned u; float f; } x; x.u = ((unsigned)y.s) << 16;
  return x.f;
}
__device__ __forceinline__ float sigmoid_f(float x) {
  return 1.0f / (1.0f + __expf(-x));
}
__device__ __forceinline__ float tanh_f(float x) {
  float e = __expf(2.0f * x);
  return 1.0f - 2.0f / (e + 1.0f);
}

// ---- fragment loads (contiguous k = 16h + s, same order for A and B) ---------------------------
// A 16x32 bf16: lane m=lane&15; halves 0-15 = K[2*koff..2*koff+15] (contiguous)
// B 32x16 bf16: lane n=lane&15; halves 0-15 = K[kgrp..kgrp+15] (contiguous)

__device__ __forceinline__ bf16x16 load_a_frag(const __bf16* row, int k, int koff) {
  bf16x16 r;
  ((uint4*)&r)[0] = *(const uint4*)(row + k + 2 * koff);
  ((uint4*)&r)[1] = *(const uint4*)(row + k + 2 * koff + 8);
  return r;
}
__device__ __forceinline__ bf16x16 load_b_frag(const __bf16* col, int k) {
  bf16x16 r;
  ((uint4*)&r)[0] = *(const uint4*)(col + k);
  ((uint4*)&r)[1] = *(const uint4*)(col + k + 8);
  return r;
}

// ---------------------------------------------------------------------------
// Kernel: f32 [K][N] -> bf16 [N][K] (transposed, so WMMA B frags are contiguous)
// ---------------------------------------------------------------------------
__global__ void lstm_pack_wt(const float* __restrict__ W, __bf16* __restrict__ Wt,
                             int K, int N) {
  int idx = blockIdx.x * blockDim.x + threadIdx.x;
  if (idx >= K * N) return;
  int n = idx / K, k = idx - n * K;
  Wt[idx] = f2bf(W[(size_t)k * N + n]);
}

// ---------------------------------------------------------------------------
// Kernel: embedding gather -> X[t*64+b][EMBD] bf16 (row = t*64 + b)
// ---------------------------------------------------------------------------
__global__ void lstm_embed(const int* __restrict__ tokens,
                           const float* __restrict__ emb,
                           __bf16* __restrict__ X) {
  int row = blockIdx.x;              // 0..MROWS-1 = t*64 + b
  int t = row >> 6, b = row & 63;
  int tok = tokens[b * TSEQ + t];
  const float* src = emb + (size_t)tok * EMBD;
  __bf16* dst = X + (size_t)row * EMBD;
  for (int e = threadIdx.x; e < EMBD; e += blockDim.x) dst[e] = f2bf(src[e]);
}

// ---------------------------------------------------------------------------
// Kernel: C[M][G4] = A[M][K] @ Bt[G4][K]^T + bias   (bf16 WMMA, f32 accum)
// 256 threads = 8 waves. WG tile 64x128; wave tile 16x64 (4 n-tiles).
// ---------------------------------------------------------------------------
__global__ __launch_bounds__(256) void lstm_gemm_xw(
    const __bf16* __restrict__ A, const __bf16* __restrict__ Bt,
    const float* __restrict__ bias, float* __restrict__ C, int K) {
  const int w = threadIdx.x >> 5, lane = threadIdx.x & 31;
  const int mt = w >> 1, nh = w & 1;
  const int row0 = blockIdx.x * 64 + mt * 16;
  const int col0 = blockIdx.y * 128 + nh * 64;
  const int m    = lane & 15;
  const int nn   = lane & 15;
  const int koff = (lane >> 4) * 8;
  const int kgrp = (lane >> 4) * 16;

  f32x8 acc[4];
  #pragma unroll
  for (int j = 0; j < 4; ++j)
    #pragma unroll
    for (int r = 0; r < 8; ++r) acc[j][r] = 0.0f;

  const __bf16* arow = A + (size_t)(row0 + m) * K;
  for (int k = 0; k < K; k += 32) {
    bf16x16 af = load_a_frag(arow, k, koff);
    #pragma unroll
    for (int j = 0; j < 4; ++j) {
      const __bf16* bcol = Bt + (size_t)(col0 + j * 16 + nn) * K + kgrp;
      bf16x16 bfr = load_b_frag(bcol, k);
      acc[j] = WMMA_BF16(af, bfr, acc[j]);
    }
  }
  const int mrow = row0 + (lane >> 4) * 8;   // C layout: M = r + 8*(lane>=16)
  #pragma unroll
  for (int j = 0; j < 4; ++j) {
    int col = col0 + j * 16 + nn;
    float bv = bias[col];
    #pragma unroll
    for (int r = 0; r < 8; ++r)
      C[(size_t)(mrow + r) * G4 + col] = acc[j][r] + bv;
  }
}

// ---------------------------------------------------------------------------
// Kernel: full-sequence LSTM recurrence (persistent, 16 WGs x 256 threads).
// Each wave owns 32 H-columns x 4 gates for a fixed 16-row m-tile; its cell
// state stays in registers for all 512 steps.  Per step: z = Zx[t] + h@U,
// gates, h written double-buffered, then atomic grid barrier on bar[t].
// ---------------------------------------------------------------------------
__global__ __launch_bounds__(256) void lstm_seq(
    const float*  __restrict__ Zx,    // [TSEQ*64][G4]  precomputed x@W + b
    const __bf16* __restrict__ Ut,    // [G4][HID]      transposed recurrent wts
    __bf16*       __restrict__ hbuf,  // [2][64][HID]   ping-pong hidden state
    __bf16*       __restrict__ seq,   // [TSEQ*64][HID] or nullptr
    int*          __restrict__ bar) { // [TSEQ] zeroed barrier counters
  const int w = threadIdx.x >> 5, lane = threadIdx.x & 31;
  const int mt = w >> 1, nh = w & 1;
  const int colbase = blockIdx.x * 64 + nh * 32;   // 2 n-tiles of H columns
  const int m    = lane & 15;
  const int nn   = lane & 15;
  const int koff = (lane >> 4) * 8;
  const int kgrp = (lane >> 4) * 16;
  const int mrow = mt * 16 + (lane >> 4) * 8;      // batch row base (C layout)
  const int stateN = BATCH * HID;

  // B-fragment base pointers, invariant across time
  const __bf16* bcol[4][2];
  #pragma unroll
  for (int g = 0; g < 4; ++g)
    #pragma unroll
    for (int j = 0; j < 2; ++j)
      bcol[g][j] = Ut + (size_t)(g * HID + colbase + j * 16 + nn) * HID + kgrp;

  // cell state in registers: lane exclusively owns (mrow+r, colbase+j*16+nn)
  float creg[2][8];
  #pragma unroll
  for (int j = 0; j < 2; ++j)
    #pragma unroll
    for (int r = 0; r < 8; ++r) creg[j][r] = 0.0f;

  for (int t = 0; t < TSEQ; ++t) {
    const __bf16* hin  = hbuf + (size_t)(t & 1) * stateN;
    __bf16*       hout = hbuf + (size_t)((t + 1) & 1) * stateN;
    const float*  Zt   = Zx + (size_t)t * BATCH * G4;

    f32x8 acc[4][2];
    #pragma unroll
    for (int g = 0; g < 4; ++g)
      #pragma unroll
      for (int j = 0; j < 2; ++j) {
        int col = g * HID + colbase + j * 16 + nn;
        #pragma unroll
        for (int r = 0; r < 8; ++r)
          acc[g][j][r] = Zt[(size_t)(mrow + r) * G4 + col];
      }

    const __bf16* arow = hin + (size_t)(mt * 16 + m) * HID;
    for (int k = 0; k < HID; k += 32) {
      bf16x16 af = load_a_frag(arow, k, koff);
      #pragma unroll
      for (int g = 0; g < 4; ++g)
        #pragma unroll
        for (int j = 0; j < 2; ++j) {
          bf16x16 bfr = load_b_frag(bcol[g][j], k);
          acc[g][j] = WMMA_BF16(af, bfr, acc[g][j]);
        }
    }

    __bf16* srow = seq ? seq + (size_t)t * BATCH * HID : nullptr;
    #pragma unroll
    for (int j = 0; j < 2; ++j) {
      int col = colbase + j * 16 + nn;
      #pragma unroll
      for (int r = 0; r < 8; ++r) {
        float iv = sigmoid_f(acc[0][j][r]);
        float fv = sigmoid_f(acc[1][j][r]);
        float gv = tanh_f(acc[2][j][r]);
        float ov = sigmoid_f(acc[3][j][r]);
        float cn = fv * creg[j][r] + iv * gv;
        creg[j][r] = cn;
        __bf16 hb = f2bf(ov * tanh_f(cn));
        size_t hi = (size_t)(mrow + r) * HID + col;
        hout[hi] = hb;
        if (srow) srow[hi] = hb;
      }
    }

    // ---- grid barrier: fresh counter per step (no reset hazard) ----
    if (t != TSEQ - 1) {
      __threadfence();                       // release h stores device-wide
      __syncthreads();
      if (threadIdx.x == 0) {
        atomicAdd(&bar[t], 1);
        while (atomicAdd(&bar[t], 0) < NWG) __builtin_amdgcn_s_sleep(2);
      }
      __syncthreads();
      __threadfence();                       // acquire other WGs' h stores
    }
  }
}

// ---------------------------------------------------------------------------
// Kernel: zero n dwords
// ---------------------------------------------------------------------------
__global__ void lstm_zero_u32(unsigned* __restrict__ p, int n) {
  int i = blockIdx.x * blockDim.x + threadIdx.x;
  if (i < n) p[i] = 0u;
}

// ---------------------------------------------------------------------------
// Kernel: out[b][cls] = softmax(h[b]@Wd + bd).  128 threads: (b, cls) pairs.
// ---------------------------------------------------------------------------
__global__ void lstm_head(const __bf16* __restrict__ h, const float* __restrict__ Wd,
                          const float* __restrict__ bd, float* __restrict__ out) {
  int tid = threadIdx.x;                // 0..127
  int b = tid >> 1, cls = tid & 1;
  const __bf16* hr = h + (size_t)b * HID;
  float acc = bd[cls];
  for (int k = 0; k < HID; ++k) acc += bf2f(hr[k]) * Wd[k * 2 + cls];
  float other = __shfl_xor(acc, 1, 32);
  float mx = fmaxf(acc, other);
  float e = __expf(acc - mx), eo = __expf(other - mx);
  out[b * 2 + cls] = e / (e + eo);
}

// ---------------------------------------------------------------------------
// Host launcher
// ---------------------------------------------------------------------------
extern "C" void kernel_launch(void* const* d_in, const int* in_sizes, int n_in,
                              void* d_out, int out_size, void* d_ws, size_t ws_size,
                              hipStream_t stream) {
  (void)in_sizes; (void)n_in; (void)out_size; (void)ws_size;
  const int*   tokens = (const int*)  d_in[0];
  const float* emb    = (const float*)d_in[1];
  const float* W0     = (const float*)d_in[2];
  const float* U0     = (const float*)d_in[3];
  const float* b0     = (const float*)d_in[4];
  const float* W1     = (const float*)d_in[5];
  const float* U1     = (const float*)d_in[6];
  const float* b1     = (const float*)d_in[7];
  const float* Wd     = (const float*)d_in[8];
  const float* bd     = (const float*)d_in[9];
  float* out = (float*)d_out;

  char* ws = (char*)d_ws;
  size_t off = 0;
  auto carve = [&](size_t bytes) -> void* {
    void* p = ws + off;
    off += (bytes + 255) & ~(size_t)255;
    return p;
  };
  float*  Zx   = (float*) carve((size_t)MROWS * G4 * sizeof(float));   // 512 MB
  __bf16* X    = (__bf16*)carve((size_t)MROWS * EMBD * 2);             //  32 MB
  __bf16* Seq  = (__bf16*)carve((size_t)MROWS * HID * 2);              //  64 MB
  __bf16* W0t  = (__bf16*)carve((size_t)G4 * EMBD * 2);
  __bf16* U0t  = (__bf16*)carve((size_t)G4 * HID * 2);
  __bf16* W1t  = (__bf16*)carve((size_t)G4 * HID * 2);
  __bf16* U1t  = (__bf16*)carve((size_t)G4 * HID * 2);
  __bf16* hbuf = (__bf16*)carve((size_t)2 * BATCH * HID * 2);          // ping/pong
  int*    bar0 = (int*)   carve((size_t)TSEQ * sizeof(int));
  int*    bar1 = (int*)   carve((size_t)TSEQ * sizeof(int));

  const int stateN = BATCH * HID;
  const int hZeroN = 2 * stateN / 2;   // 2*stateN bf16 = stateN u32 dwords... (bytes/4)
  // 2 * stateN * sizeof(bf16) bytes = 2*65536*2 = 262144 B = 65536 dwords
  const int hDwords = (2 * stateN * 2) / 4;

  // 1) weight packing (f32 -> bf16, [K][N] -> [N][K])
  lstm_pack_wt<<<(G4 * EMBD + 255) / 256, 256, 0, stream>>>(W0, W0t, EMBD, G4);
  lstm_pack_wt<<<(G4 * HID  + 255) / 256, 256, 0, stream>>>(U0, U0t, HID,  G4);
  lstm_pack_wt<<<(G4 * HID  + 255) / 256, 256, 0, stream>>>(W1, W1t, HID,  G4);
  lstm_pack_wt<<<(G4 * HID  + 255) / 256, 256, 0, stream>>>(U1, U1t, HID,  G4);

  // 2) embedding gather
  lstm_embed<<<MROWS, 256, 0, stream>>>(tokens, emb, X);

  // 3) layer-0 input projection: Zx = X @ W0 + b0   [32768, 4096]
  {
    dim3 grid(MROWS / 64, G4 / 128);
    lstm_gemm_xw<<<grid, 256, 0, stream>>>(X, W0t, b0, Zx, EMBD);
  }

  // 4) layer-0 recurrence (single persistent launch)
  lstm_zero_u32<<<(hDwords + 255) / 256, 256, 0, stream>>>((unsigned*)hbuf, hDwords);
  lstm_zero_u32<<<(TSEQ + 255) / 256, 256, 0, stream>>>((unsigned*)bar0, TSEQ);
  lstm_seq<<<NWG, 256, 0, stream>>>(Zx, U0t, hbuf, Seq, bar0);

  // 5) layer-1 input projection: Zx = Seq @ W1 + b1
  {
    dim3 grid(MROWS / 64, G4 / 128);
    lstm_gemm_xw<<<grid, 256, 0, stream>>>(Seq, W1t, b1, Zx, HID);
  }

  // 6) layer-1 recurrence
  lstm_zero_u32<<<(hDwords + 255) / 256, 256, 0, stream>>>((unsigned*)hbuf, hDwords);
  lstm_zero_u32<<<(TSEQ + 255) / 256, 256, 0, stream>>>((unsigned*)bar1, TSEQ);
  lstm_seq<<<NWG, 256, 0, stream>>>(Zx, U1t, hbuf, nullptr, bar1);

  // final h is in hbuf[(TSEQ)&1 == 0] half
  lstm_head<<<1, 128, 0, stream>>>(hbuf, Wd, bd, out);
  (void)hZeroN;
}
